// MultiLatentAttention_26276609917414
// MI455X (gfx1250) — compile-verified, matrix operand order checked by reading, not run on hardware
//
#include <hip/hip_runtime.h>
#include <hip/hip_bf16.h>

typedef __attribute__((ext_vector_type(16))) _Float16 v16h;
typedef __attribute__((ext_vector_type(8)))  _Float16 v8h;
typedef __attribute__((ext_vector_type(8)))  float    v8f;
typedef __attribute__((ext_vector_type(4)))  unsigned u32x4;
typedef __attribute__((ext_vector_type(8)))  unsigned u32x8;

#define SEQ     2048
#define HDIM    768
#define NHEADS  12
#define HEADD   64
#define D2      128      // 2*HEADD, attention feature dim
#define LRDIM   384
#define WIN     1024
#define EOT_TOK 50256

// ---------------------------------------------------------------------------
// WMMA fragment loader (A or B^T). Row-major tile base, leading dim `ld`
// (elements). Per §7.12.2 16-bit A layout: lane L holds row L&15; K elements
// are {klo..klo+7, 16+klo..16+klo+7}, klo = (L>=16)?8:0. Two 16-byte loads.
// Works for global or LDS-backed (generic) pointers.
// ---------------------------------------------------------------------------
__device__ __forceinline__ v16h load_frag(const _Float16* __restrict__ tileBase,
                                          int ld, int k0, int lane) {
    int r   = lane & 15;
    int klo = (lane & 16) ? 8 : 0;
    const _Float16* p = tileBase + (size_t)r * ld + k0 + klo;
    v8h lo = *(const v8h*)p;
    v8h hi = *(const v8h*)(p + 16);
    return __builtin_shufflevector(lo, hi, 0,1,2,3,4,5,6,7,8,9,10,11,12,13,14,15);
}

// ---------------------------------------------------------------------------
// TDM: DMA a 2D f16 tile (rows x rowElems, global leading dim strideElems)
// into LDS at byte-offset ldsOff. D# per CDNA5 ISA §8.3/8.4:
//   group0: count=1 | lds_addr | global_addr | type=2
//   group1: data_size=1 (2B) | tensor_dim0/1 | tile_dim0/1 | dim0_stride
// 2D tile -> only groups 0/1 (VADDR2/3 omitted). Tracked by TENSORcnt.
// ---------------------------------------------------------------------------
__device__ __forceinline__ void tdm_load_2d(unsigned ldsOff, const void* gptr,
                                            unsigned rowElems, unsigned rows,
                                            unsigned strideElems) {
    unsigned long long ga = (unsigned long long)(uintptr_t)gptr;
    u32x4 g0;
    g0[0] = 1u;                                   // count=1, user mode
    g0[1] = ldsOff;                               // lds_addr [63:32]
    g0[2] = (unsigned)ga;                         // global_addr low
    g0[3] = (unsigned)(ga >> 32) | 0x80000000u;   // global_addr hi | type=2
    u32x8 g1;
    g1[0] = 0x00010000u;                          // data_size=1 (2 bytes)
    g1[1] = (rowElems & 0xFFFFu) << 16;           // tensor_dim0[15:0]
    g1[2] = ((rowElems >> 16) & 0xFFFFu)          // tensor_dim0[31:16]
          | ((rows & 0xFFFFu) << 16);             // tensor_dim1[15:0]
    g1[3] = ((rows >> 16) & 0xFFFFu)              // tensor_dim1[31:16]
          | ((rowElems & 0xFFFFu) << 16);         // tile_dim0
    g1[4] = rows & 0xFFFFu;                       // tile_dim1 (tile_dim2=0)
    g1[5] = strideElems;                          // tensor_dim0_stride[31:0]
    g1[6] = 0u;                                   // stride hi / dim1_stride lo
    g1[7] = 0u;
    asm volatile("tensor_load_to_lds %0, %1" :: "s"(g0), "s"(g1) : "memory");
}

// ---------------------------------------------------------------------------
// Register-blocked f16 WMMA GEMM: each wave owns a 32x64 strip of C
// (2 M-tiles x 4 N-tiles): per k-step 8 WMMAs on 2 A-frag + 4 B-frag loads.
//   A : f16 row-major (lda), Wt: f16 N x K row-major (W transposed, ldb),
//   C : OutT row-major (ldc), bias f32[N]. 8 waves / block.
// ---------------------------------------------------------------------------
template <typename OutT>
__global__ __launch_bounds__(256) void gemm_blk(const _Float16* __restrict__ A, int lda,
                                                const _Float16* __restrict__ Wt, int ldb,
                                                const float* __restrict__ bias,
                                                OutT* __restrict__ C, int ldc,
                                                int M, int N, int K) {
    int lane   = threadIdx.x & 31;
    int wave   = threadIdx.x >> 5;
    int stripsN = N >> 6;                         // 64-wide strips
    int strip  = blockIdx.x * 8 + wave;
    if (strip >= (M >> 5) * stripsN) return;      // whole-wave exit only
    int sm = strip / stripsN;
    int sn = strip % stripsN;

    const _Float16* A0 = A  + (size_t)sm * 32 * lda;
    const _Float16* B0 = Wt + (size_t)sn * 64 * ldb;

    v8f acc[2][4];
#pragma unroll
    for (int i = 0; i < 2; ++i)
#pragma unroll
        for (int j = 0; j < 4; ++j) acc[i][j] = (v8f){0,0,0,0,0,0,0,0};

    for (int k0 = 0; k0 < K; k0 += 32) {
        v16h a0 = load_frag(A0,            lda, k0, lane);
        v16h a1 = load_frag(A0 + 16 * lda, lda, k0, lane);
#pragma unroll
        for (int j = 0; j < 4; ++j) {
            v16h b = load_frag(B0 + (size_t)j * 16 * ldb, ldb, k0, lane);
            acc[0][j] = __builtin_amdgcn_wmma_f32_16x16x32_f16(false, a0, false, b,
                                                               (short)0, acc[0][j], false, false);
            acc[1][j] = __builtin_amdgcn_wmma_f32_16x16x32_f16(false, a1, false, b,
                                                               (short)0, acc[1][j], false, false);
        }
    }

    int cl = lane & 15;
    int rb = (lane & 16) ? 8 : 0;
#pragma unroll
    for (int j = 0; j < 4; ++j) {
        int   col = sn * 64 + j * 16 + cl;
        float bv  = bias[col];
#pragma unroll
        for (int i = 0; i < 2; ++i) {
            int rowb = sm * 32 + i * 16 + rb;
#pragma unroll
            for (int r = 0; r < 8; ++r)
                C[(size_t)(rowb + r) * ldc + col] = (OutT)(acc[i][j][r] + bv);
        }
    }
}

// ---------------------------------------------------------------------------
// Small helpers
// ---------------------------------------------------------------------------
__global__ void cvt_f16(const float* __restrict__ in, _Float16* __restrict__ out, int n) {
    int i = blockIdx.x * blockDim.x + threadIdx.x;
    if (i < n) out[i] = (_Float16)in[i];
}

__global__ void transpose_f16(const float* __restrict__ w, _Float16* __restrict__ wt,
                              int K, int N) {
    int i = blockIdx.x * blockDim.x + threadIdx.x;
    if (i >= K * N) return;
    int k = i / N, n = i % N;
    wt[(size_t)n * K + k] = (_Float16)w[i];
}

__global__ void copy_f32(const float* __restrict__ in, float* __restrict__ out, int n) {
    int i = blockIdx.x * blockDim.x + threadIdx.x;
    if (i < n) out[i] = in[i];
}

__global__ void docs_scan(const int* __restrict__ tok, int* __restrict__ docs, int n) {
    if (blockIdx.x == 0 && threadIdx.x == 0) {
        int c = 0;
        for (int i = 0; i < n; ++i) { c += (tok[i] == EOT_TOK); docs[i] = c; }
    }
}

// ---------------------------------------------------------------------------
// Assemble: RoPE(qr/kr) + pack Qf/Kf as [head][S][128] and Vt as [head][128][S]
// ---------------------------------------------------------------------------
__global__ __launch_bounds__(128) void assemble(const _Float16* __restrict__ lat,
                                                const _Float16* __restrict__ qbuf,
                                                const _Float16* __restrict__ kvbuf,
                                                _Float16* __restrict__ Qf,
                                                _Float16* __restrict__ Kf,
                                                _Float16* __restrict__ Vt) {
    int s = blockIdx.x;
    int d = threadIdx.x;          // 0..127
    _Float16 qr_r = (_Float16)0.f, kr_r = (_Float16)0.f;
    if (d < HEADD) {
        int   i   = d & 31;       // rotary pair index
        float div = __expf(-__logf(10000.f) * (2.f * (float)i) / (float)HEADD);
        float ang = (float)s * div;
        float sn = __sinf(ang), cs = __cosf(ang);
        const _Float16* qr = lat + (size_t)s * 896 + 768;
        const _Float16* kr = lat + (size_t)s * 896 + 832;
        float xq = (float)qr[d];
        float xk = (float)kr[d];
        float rq = (d < 32) ? -(float)qr[d + 32] : (float)qr[d - 32];
        float rk = (d < 32) ? -(float)kr[d + 32] : (float)kr[d - 32];
        qr_r = (_Float16)(xq * cs + rq * sn);
        kr_r = (_Float16)(xk * cs + rk * sn);
    }
#pragma unroll
    for (int h = 0; h < NHEADS; ++h) {
        size_t base = ((size_t)h * SEQ + s) * D2;
        if (d < HEADD) {
            Qf[base + d]         = qbuf[(size_t)s * HDIM + h * HEADD + d];
            Qf[base + HEADD + d] = qr_r;
            Kf[base + d]         = kvbuf[(size_t)s * 2304 + h * HEADD + d];
            Kf[base + HEADD + d] = kr_r;
        }
        Vt[((size_t)h * D2 + d) * SEQ + s] = kvbuf[(size_t)s * 2304 + HDIM + h * D2 + d];
    }
}

// ---------------------------------------------------------------------------
// Flash attention, window-aware, TDM double-buffered K/V staging in LDS.
// 4 waves/block; one wave per (head, 16-row q tile). Per 32-wide kv step:
// 8 score WMMAs + 8 PV WMMAs, K/V tiles DMA'd by the Tensor Data Mover.
// ---------------------------------------------------------------------------
__global__ __launch_bounds__(128) void flash_attn(const _Float16* __restrict__ Qf,
                                                  const _Float16* __restrict__ Kf,
                                                  const _Float16* __restrict__ Vt,
                                                  const int* __restrict__ docs,
                                                  _Float16* __restrict__ attn_out) {
    __shared__ __align__(16) _Float16 kbuf[4][2][32][D2];  // 64 KB
    __shared__ __align__(16) _Float16 vbuf[4][2][D2][32];  // 64 KB
    __shared__ __align__(16) _Float16 pbuf[4][16][32];     //  4 KB

    int lane = threadIdx.x & 31;
    int wave = threadIdx.x >> 5;
    int wid  = blockIdx.x * 4 + wave;
    if (wid >= NHEADS * (SEQ / 16)) return;               // whole-wave exit
    int head = wid >> 7;                                   // / 128 q-tiles
    int q0   = (wid & 127) * 16;

    const float scale = 0.08838834764831845f;             // 1/sqrt(128)
    const float L2E   = 1.44269504088896f;

    int col  = lane & 15;
    int half = lane >> 4;
    int klo  = half ? 8 : 0;

    const _Float16* Qbase  = Qf + ((size_t)head * SEQ + q0) * D2;
    const _Float16* KfBase = Kf + (size_t)head * SEQ * D2;
    const _Float16* VtBase = Vt + (size_t)head * D2 * SEQ;

    v16h qfrag[4];
#pragma unroll
    for (int kk = 0; kk < 4; ++kk) qfrag[kk] = load_frag(Qbase, D2, kk * 32, lane);

    v8f O[8];
    float mrow[8], lrow[8];
    int  qrow[8], qdoc[8];
#pragma unroll
    for (int t = 0; t < 8; ++t) O[t] = (v8f){0,0,0,0,0,0,0,0};
#pragma unroll
    for (int r = 0; r < 8; ++r) {
        mrow[r] = -1e30f; lrow[r] = 0.f;
        qrow[r] = q0 + r + 8 * half;
        qdoc[r] = docs[qrow[r]];
    }

    int kvStart = q0 + 15 - (WIN - 1);
    if (kvStart < 0) kvStart = 0;
    kvStart &= ~31;
    int kvEnd = q0 + 15;

    // prologue: stage first K/V tiles into buffer 0 via TDM
    unsigned kOff0 = (unsigned)(uintptr_t)(void*)&kbuf[wave][0][0][0];
    unsigned kOff1 = (unsigned)(uintptr_t)(void*)&kbuf[wave][1][0][0];
    unsigned vOff0 = (unsigned)(uintptr_t)(void*)&vbuf[wave][0][0][0];
    unsigned vOff1 = (unsigned)(uintptr_t)(void*)&vbuf[wave][1][0][0];
    tdm_load_2d(kOff0, KfBase + (size_t)kvStart * D2, D2, 32, D2);
    tdm_load_2d(vOff0, VtBase + kvStart,              32, D2, SEQ);

    int buf = 0;
    for (int kv0 = kvStart; kv0 <= kvEnd; kv0 += 32) {
        int kvNext = kv0 + 32;
        if (kvNext <= kvEnd) {      // prefetch next tiles into the other buffer
            tdm_load_2d(buf ? kOff0 : kOff1, KfBase + (size_t)kvNext * D2, D2, 32, D2);
            tdm_load_2d(buf ? vOff0 : vOff1, VtBase + kvNext,              32, D2, SEQ);
            __builtin_amdgcn_s_wait_tensorcnt(2);   // in-order: current pair done
        } else {
            __builtin_amdgcn_s_wait_tensorcnt(0);
        }
        asm volatile("" ::: "memory");

        const _Float16* Kb = &kbuf[wave][buf][0][0];     // [32][128]
        const _Float16* Vb = &vbuf[wave][buf][0][0];     // [128][32]

        // ---- scores: two 16x16 tiles over the 32 kv columns ----
        v8f s0 = {0,0,0,0,0,0,0,0};
        v8f s1 = {0,0,0,0,0,0,0,0};
#pragma unroll
        for (int kk = 0; kk < 4; ++kk) {
            v16h kf0 = load_frag(Kb,            D2, kk * 32, lane);
            v16h kf1 = load_frag(Kb + 16 * D2,  D2, kk * 32, lane);
            s0 = __builtin_amdgcn_wmma_f32_16x16x32_f16(false, qfrag[kk], false, kf0,
                                                        (short)0, s0, false, false);
            s1 = __builtin_amdgcn_wmma_f32_16x16x32_f16(false, qfrag[kk], false, kf1,
                                                        (short)0, s1, false, false);
        }

        // ---- mask + online softmax ----
        int kc0 = kv0 + col, kc1 = kv0 + 16 + col;
        int kd0 = docs[kc0], kd1 = docs[kc1];
#pragma unroll
        for (int r = 0; r < 8; ++r) {
            int  qi = qrow[r];
            bool m0 = (qi >= kc0) && (qi - kc0 < WIN) && (qdoc[r] == kd0);
            bool m1 = (qi >= kc1) && (qi - kc1 < WIN) && (qdoc[r] == kd1);
            float v0 = m0 ? s0[r] * scale : -1e30f;
            float v1 = m1 ? s1[r] * scale : -1e30f;
            float mx = fmaxf(v0, v1);
#pragma unroll
            for (int off = 1; off < 16; off <<= 1)
                mx = fmaxf(mx, __shfl_xor(mx, off, 32));
            float mnew  = fmaxf(mrow[r], mx);
            float alpha = exp2f((mrow[r] - mnew) * L2E);
            mrow[r] = mnew;
            float p0 = exp2f((v0 - mnew) * L2E);
            float p1 = exp2f((v1 - mnew) * L2E);
            float rs = p0 + p1;
#pragma unroll
            for (int off = 1; off < 16; off <<= 1)
                rs += __shfl_xor(rs, off, 32);
            lrow[r] = lrow[r] * alpha + rs;
#pragma unroll
            for (int t = 0; t < 8; ++t) O[t][r] *= alpha;
            int prow = r + 8 * half;
            pbuf[wave][prow][col]      = (_Float16)p0;
            pbuf[wave][prow][col + 16] = (_Float16)p1;
        }

        // wave-private LDS round-trip: C-layout -> A-fragment layout
        asm volatile("s_wait_dscnt 0x0" ::: "memory");
        v8h plo = *(const v8h*)&pbuf[wave][col][klo];
        v8h phi = *(const v8h*)&pbuf[wave][col][16 + klo];
        v16h pfrag = __builtin_shufflevector(plo, phi,
                        0,1,2,3,4,5,6,7,8,9,10,11,12,13,14,15);

        // ---- O += P @ V (V tile staged transposed: [128][32]) ----
#pragma unroll
        for (int t = 0; t < 8; ++t) {
            v16h vfrag = load_frag(Vb + (size_t)t * 16 * 32, 32, 0, lane);
            O[t] = __builtin_amdgcn_wmma_f32_16x16x32_f16(false, pfrag, false, vfrag,
                                                          (short)0, O[t], false, false);
        }
        buf ^= 1;
    }

    // ---- epilogue: normalize, store f16 [S][1536] (col = head*128 + d) ----
#pragma unroll
    for (int r = 0; r < 8; ++r) {
        float inv = 1.0f / lrow[r];
        int   row = qrow[r];
#pragma unroll
        for (int t = 0; t < 8; ++t)
            attn_out[(size_t)row * (NHEADS * D2) + head * D2 + t * 16 + col] =
                (_Float16)(O[t][r] * inv);
    }
}

// ---------------------------------------------------------------------------
// Host-side orchestration
// ---------------------------------------------------------------------------
extern "C" void kernel_launch(void* const* d_in, const int* in_sizes, int n_in,
                              void* d_out, int out_size, void* d_ws, size_t ws_size,
                              hipStream_t stream) {
    (void)in_sizes; (void)n_in; (void)out_size; (void)ws_size;
    const float* x     = (const float*)d_in[0];
    const int*   tok   = (const int*)  d_in[1];
    const float* qd_w  = (const float*)d_in[2];
    const float* qd_b  = (const float*)d_in[3];
    const float* qu_w  = (const float*)d_in[4];
    const float* qu_b  = (const float*)d_in[5];
    const float* qr_w  = (const float*)d_in[6];
    const float* qr_b  = (const float*)d_in[7];
    const float* kvd_w = (const float*)d_in[8];
    const float* kvd_b = (const float*)d_in[9];
    const float* ku_w  = (const float*)d_in[10];
    const float* ku_b  = (const float*)d_in[11];
    const float* vu_w  = (const float*)d_in[12];
    const float* vu_b  = (const float*)d_in[13];
    const float* kr_w  = (const float*)d_in[14];
    const float* kr_b  = (const float*)d_in[15];
    const float* o_w   = (const float*)d_in[16];
    const float* o_b   = (const float*)d_in[17];
    float* out = (float*)d_out;

    char* ws = (char*)d_ws;
    size_t off = 0;
    auto carve = [&](size_t bytes) -> char* {
        char* p = ws + off;
        off = (off + bytes + 255) & ~(size_t)255;
        return p;
    };
    _Float16* xb      = (_Float16*)carve((size_t)SEQ * HDIM * 2);
    _Float16* wt1     = (_Float16*)carve((size_t)896 * HDIM * 2);
    float*    b1      = (float*)   carve(896 * 4);
    _Float16* lat     = (_Float16*)carve((size_t)SEQ * 896 * 2);
    _Float16* wt_qu   = (_Float16*)carve((size_t)HDIM * LRDIM * 2);
    _Float16* qbuf    = (_Float16*)carve((size_t)SEQ * HDIM * 2);
    _Float16* wt_kv   = (_Float16*)carve((size_t)2304 * LRDIM * 2);
    float*    bkv     = (float*)   carve(2304 * 4);
    _Float16* kvbuf   = (_Float16*)carve((size_t)SEQ * 2304 * 2);
    int*      docs    = (int*)     carve(SEQ * 4);
    _Float16* Qf      = (_Float16*)carve((size_t)NHEADS * SEQ * D2 * 2);
    _Float16* Kf      = (_Float16*)carve((size_t)NHEADS * SEQ * D2 * 2);
    _Float16* Vt      = (_Float16*)carve((size_t)NHEADS * D2 * SEQ * 2);
    _Float16* attnout = (_Float16*)carve((size_t)SEQ * NHEADS * D2 * 2);
    _Float16* wt_o    = (_Float16*)carve((size_t)HDIM * 1536 * 2);

    auto cdiv = [](int a, int b) { return (a + b - 1) / b; };

    cvt_f16<<<cdiv(SEQ * HDIM, 256), 256, 0, stream>>>(x, xb, SEQ * HDIM);
    transpose_f16<<<cdiv(HDIM * LRDIM, 256), 256, 0, stream>>>(qd_w,  wt1,                    HDIM, LRDIM);
    transpose_f16<<<cdiv(HDIM * LRDIM, 256), 256, 0, stream>>>(kvd_w, wt1 + (size_t)384*HDIM, HDIM, LRDIM);
    transpose_f16<<<cdiv(HDIM * 64, 256),    256, 0, stream>>>(qr_w,  wt1 + (size_t)768*HDIM, HDIM, 64);
    transpose_f16<<<cdiv(HDIM * 64, 256),    256, 0, stream>>>(kr_w,  wt1 + (size_t)832*HDIM, HDIM, 64);
    transpose_f16<<<cdiv(LRDIM * HDIM, 256), 256, 0, stream>>>(qu_w,  wt_qu, LRDIM, HDIM);
    transpose_f16<<<cdiv(LRDIM * HDIM, 256), 256, 0, stream>>>(ku_w,  wt_kv, LRDIM, HDIM);
    transpose_f16<<<cdiv(LRDIM * 1536, 256), 256, 0, stream>>>(vu_w,  wt_kv + (size_t)768*LRDIM, LRDIM, 1536);
    transpose_f16<<<cdiv(1536 * HDIM, 256),  256, 0, stream>>>(o_w,   wt_o, 1536, HDIM);
    copy_f32<<<cdiv(384, 256), 256, 0, stream>>>(qd_b,  b1,       384);
    copy_f32<<<cdiv(384, 256), 256, 0, stream>>>(kvd_b, b1 + 384, 384);
    copy_f32<<<1, 64, 0, stream>>>(qr_b, b1 + 768, 64);
    copy_f32<<<1, 64, 0, stream>>>(kr_b, b1 + 832, 64);
    copy_f32<<<cdiv(768, 256),  256, 0, stream>>>(ku_b, bkv,       768);
    copy_f32<<<cdiv(1536, 256), 256, 0, stream>>>(vu_b, bkv + 768, 1536);
    docs_scan<<<1, 1, 0, stream>>>(tok, docs, SEQ);

    // stage 1: x @ [qd|kvd|qr|kr] -> lat (S x 896)
    {
        int strips = (SEQ / 32) * (896 / 64);
        gemm_blk<_Float16><<<cdiv(strips, 8), 256, 0, stream>>>(
            xb, HDIM, wt1, HDIM, b1, lat, 896, SEQ, 896, HDIM);
    }
    // stage 2a: lq @ qu -> qbuf (S x 768)
    {
        int strips = (SEQ / 32) * (HDIM / 64);
        gemm_blk<_Float16><<<cdiv(strips, 8), 256, 0, stream>>>(
            lat, 896, wt_qu, LRDIM, qu_b, qbuf, HDIM, SEQ, HDIM, LRDIM);
    }
    // stage 2b: lkv @ [ku|vu] -> kvbuf (S x 2304)
    {
        int strips = (SEQ / 32) * (2304 / 64);
        gemm_blk<_Float16><<<cdiv(strips, 8), 256, 0, stream>>>(
            lat + 384, 896, wt_kv, LRDIM, bkv, kvbuf, 2304, SEQ, 2304, LRDIM);
    }

    assemble<<<SEQ, 128, 0, stream>>>(lat, qbuf, kvbuf, Qf, Kf, Vt);

    flash_attn<<<(NHEADS * (SEQ / 16)) / 4, 128, 0, stream>>>(Qf, Kf, Vt, docs, attnout);

    // output projection: attn_out @ o_w + o_b -> f32 d_out (S x 768)
    {
        int strips = (SEQ / 32) * (HDIM / 64);
        gemm_blk<float><<<cdiv(strips, 8), 256, 0, stream>>>(
            attnout, NHEADS * D2, wt_o, 1536, o_b, out, HDIM, SEQ, HDIM, 1536);
    }
}
